// MultiheadAttention_84172769068201
// MI455X (gfx1250) — matrix operand order repaired and checked by reading, not compiled, not run on hardware
//
#include <hip/hip_runtime.h>
#include <math.h>

// ---------------------------------------------------------------------------
// Problem constants (from reference): B=2, L=2048, E=1024, H=8, D=64, HID=512
// ---------------------------------------------------------------------------
#define B_    2
#define L_    2048
#define E_    1024
#define H_    8
#define D_    64
#define HID_  512
#define ROWS_ (B_ * L_)   // 4096

typedef __attribute__((ext_vector_type(16))) __bf16 v16bf;
typedef __attribute__((ext_vector_type(8)))  float  v8f;

// A-fragment assembly: two adjacent 16B runs (k = 16h..16h+15) -> one 32B v16bf register pair
union AFrag { float4 f4[2]; v16bf v; };
union PFrag { __bf16 e[16]; v16bf v; };

static __device__ __forceinline__ v8f wmma_bf16(v16bf a, v16bf b, v8f c) {
  // (neg_a, A, neg_b, B, c_mod, C, reuse_a, reuse_b)
  return __builtin_amdgcn_wmma_f32_16x16x32_bf16(false, a, false, b, (short)0, c,
                                                 false, false);
}

static __device__ __forceinline__ v8f v8f_zero() {
  v8f z = {0.f, 0.f, 0.f, 0.f, 0.f, 0.f, 0.f, 0.f};
  return z;
}

// ---------------------------------------------------------------------------
// Prep: fp32 -> bf16 row-major copy (for the X input matrix)
// ---------------------------------------------------------------------------
__global__ void k_cvt_bf16(const float* __restrict__ x, __bf16* __restrict__ y, int n) {
  int i = blockIdx.x * blockDim.x + threadIdx.x;
  if (i < n) y[i] = (__bf16)x[i];
}

// ---------------------------------------------------------------------------
// Prep: W [K][N] fp32 row-major  ->  Wt [N][K] bf16  (K-major = WMMA B layout)
// ---------------------------------------------------------------------------
__global__ void k_transpose_bf16(const float* __restrict__ w, __bf16* __restrict__ wt,
                                 int K, int N) {
  __shared__ float t[16][17];
  int n = blockIdx.x * 16 + threadIdx.x;
  int k = blockIdx.y * 16 + threadIdx.y;
  t[threadIdx.y][threadIdx.x] = w[k * N + n];
  __syncthreads();
  int no = blockIdx.x * 16 + threadIdx.y;
  int ko = blockIdx.y * 16 + threadIdx.x;
  wt[no * K + ko] = (__bf16)t[threadIdx.x][threadIdx.y];
}

// ---------------------------------------------------------------------------
// Fused QKVG GEMM: Y = X @ [Wqk | Wvg] + bias, then
//   q -> bf16 [row][0..511]          (row-major, A-layout-friendly)
//   k -> bf16 [row][0..511]          (d-contiguous, B-layout-friendly)
//   tanh(v) -> bf16 tv_t[b][h][d][l] (l-contiguous, B-layout for PV matmul)
//   sigmoid(g) -> f32 sg[row][0..511]
// grid: (ROWS/16, 8), block 128 (4 waves x 4 N-tiles of 16 = 256 N per WG)
// ---------------------------------------------------------------------------
__global__ void k_qkvg(const __bf16* __restrict__ Xb,
                       const __bf16* __restrict__ Wqk_t,
                       const __bf16* __restrict__ Wvg_t,
                       const float*  __restrict__ bqk,
                       const float*  __restrict__ bvg,
                       __bf16* __restrict__ q_bf,
                       __bf16* __restrict__ k_bf,
                       __bf16* __restrict__ tv_t,
                       float*  __restrict__ sg) {
  const int lane = threadIdx.x & 31;
  const int wave = threadIdx.x >> 5;
  const int half = lane >> 4;
  const int l16  = lane & 15;
  const int arow = blockIdx.x * 16 + l16;
  const bool is_vg = (blockIdx.y >= 4);
  const __bf16* Wt = is_vg ? Wvg_t : Wqk_t;
  const int nloc = (blockIdx.y & 3) * 256 + wave * 64;  // within 1024-wide half

  v8f acc[4];
#pragma unroll
  for (int t = 0; t < 4; ++t) acc[t] = v8f_zero();

  for (int kc = 0; kc < E_; kc += 32) {
    AFrag a;
    a.f4[0] = *(const float4*)&Xb[arow * E_ + kc + half * 16];
    a.f4[1] = *(const float4*)&Xb[arow * E_ + kc + half * 16 + 8];
#pragma unroll
    for (int t = 0; t < 4; ++t) {
      int n = nloc + t * 16 + l16;
      v16bf bm = *(const v16bf*)&Wt[n * E_ + kc + half * 16];
      acc[t] = wmma_bf16(a.v, bm, acc[t]);
    }
  }

#pragma unroll
  for (int t = 0; t < 4; ++t) {
    int ng = blockIdx.y * 256 + wave * 64 + t * 16 + l16;  // 0..2047
    float bias = (ng < 1024) ? bqk[ng] : bvg[ng - 1024];
#pragma unroll
    for (int v = 0; v < 8; ++v) {
      float val = acc[t][v] + bias;
      int m   = v + 8 * half;
      int row = blockIdx.x * 16 + m;
      if (ng < 512) {                               // q
        q_bf[row * HID_ + ng] = (__bf16)val;
      } else if (ng < 1024) {                       // k
        k_bf[row * HID_ + (ng - 512)] = (__bf16)val;
      } else if (ng < 1536) {                       // tanh(v), transposed
        int c = ng - 1024, h = c >> 6, d = c & 63;
        int b = row >> 11, l = row & (L_ - 1);
        tv_t[((b * H_ + h) * D_ + d) * L_ + l] = (__bf16)tanhf(val);
      } else {                                      // sigmoid(g)
        sg[row * HID_ + (ng - 1536)] = 1.0f / (1.0f + __expf(-val));
      }
    }
  }
}

// ---------------------------------------------------------------------------
// Fused attention: per 16-row i-tile, 8 waves (one per head).
//   scores (WMMA) -> LDS[i][j][h] -> head-axis softmax * attn_mask * kpm
//   -> P (A-layout) -> attn += P @ tv (WMMA) -> gate with sigmoid(g) -> bf16
// attn_mask streamed exactly once from HBM; no L x L tensor materialized.
// grid: ROWS/16 = 256, block 256 (8 waves)
// ---------------------------------------------------------------------------
__global__ void k_attn(const __bf16* __restrict__ q_bf,
                       const __bf16* __restrict__ k_bf,
                       const __bf16* __restrict__ tv_t,
                       const float*  __restrict__ sg,
                       const float*  __restrict__ am,        // [B,H,L,L] f32
                       const unsigned char* __restrict__ kpm, // [B,L,L] bool
                       __bf16* __restrict__ gated) {
  __shared__ float lds[16 * 32 * 8];  // [i(16)][j(32)][h(8)]

  const int lane = threadIdx.x & 31;
  const int h    = threadIdx.x >> 5;   // wave id == head
  const int half = lane >> 4;
  const int l16  = lane & 15;
  const int ibase = blockIdx.x * 16;   // global flat row base
  const int b  = ibase >> 11;
  const int il = ibase & (L_ - 1);     // i within L

  // Q fragments for K-chunks dc=0 and dc=32 (loaded once per i-tile)
  AFrag aq0, aq1;
  {
    const __bf16* qp = &q_bf[(ibase + l16) * HID_ + h * D_];
    aq0.f4[0] = *(const float4*)&qp[half * 16];
    aq0.f4[1] = *(const float4*)&qp[half * 16 + 8];
    aq1.f4[0] = *(const float4*)&qp[32 + half * 16];
    aq1.f4[1] = *(const float4*)&qp[32 + half * 16 + 8];
  }

  v8f acc[4];
#pragma unroll
  for (int c = 0; c < 4; ++c) acc[c] = v8f_zero();
  const float scale = 0.125f;  // D^-0.5 = 64^-0.5

  for (int jc = 0; jc < L_; jc += 32) {
    // ---- scores for two 16-wide j tiles, this head ----
#pragma unroll
    for (int s = 0; s < 2; ++s) {
      int j0 = jc + s * 16;
      const __bf16* kp = &k_bf[(b * L_ + j0 + l16) * HID_ + h * D_];
      v16bf bk0 = *(const v16bf*)&kp[half * 16];
      v16bf bk1 = *(const v16bf*)&kp[32 + half * 16];
      v8f sacc = v8f_zero();
      sacc = wmma_bf16(aq0.v, bk0, sacc);
      sacc = wmma_bf16(aq1.v, bk1, sacc);
#pragma unroll
      for (int v = 0; v < 8; ++v) {
        int m = v + 8 * half;
        lds[(m * 32 + s * 16 + l16) * 8 + h] = sacc[v] * scale;
      }
    }
    __syncthreads();

    // ---- softmax over the 8 heads + both masks, in place ----
    for (int p = (int)threadIdx.x; p < 512; p += 256) {
      float* sp = &lds[p * 8];
      float4 x0 = *(float4*)sp;
      float4 x1 = *(float4*)(sp + 4);
      float xs[8] = {x0.x, x0.y, x0.z, x0.w, x1.x, x1.y, x1.z, x1.w};
      float mx = xs[0];
#pragma unroll
      for (int t = 1; t < 8; ++t) mx = fmaxf(mx, xs[t]);
      float ex[8], sum = 0.f;
#pragma unroll
      for (int t = 0; t < 8; ++t) { ex[t] = __expf(xs[t] - mx); sum += ex[t]; }
      float inv = 1.0f / sum;
      int i_l = p >> 5, j_l = p & 31;
      int ig = il + i_l, jg = jc + j_l;
      float km = kpm[((long)b * L_ + ig) * L_ + jg] ? 1.0f : 0.0f;
      const float* amp = &am[(((long)b * H_) * L_ + ig) * (long)L_ + jg];
#pragma unroll
      for (int t = 0; t < 8; ++t)
        xs[t] = ex[t] * inv * km * amp[(long)t * L_ * L_];
      *(float4*)sp       = make_float4(xs[0], xs[1], xs[2], xs[3]);
      *(float4*)(sp + 4) = make_float4(xs[4], xs[5], xs[6], xs[7]);
    }
    __syncthreads();

    // ---- read probabilities back in WMMA A layout, accumulate P @ tv ----
    PFrag P;
#pragma unroll
    for (int t = 0; t < 16; ++t) {
      int jl = half * 16 + t;
      P.e[t] = (__bf16)lds[(l16 * 32 + jl) * 8 + h];
    }
#pragma unroll
    for (int c = 0; c < 4; ++c) {
      v16bf btv = *(const v16bf*)
          &tv_t[((b * H_ + h) * D_ + c * 16 + l16) * L_ + jc + half * 16];
      acc[c] = wmma_bf16(P.v, btv, acc[c]);
    }
    __syncthreads();  // protect LDS before next iteration's score writes
  }

  // ---- gate with sigmoid(g) and emit bf16 rows for the output GEMM ----
#pragma unroll
  for (int c = 0; c < 4; ++c) {
#pragma unroll
    for (int v = 0; v < 8; ++v) {
      int m   = v + 8 * half;
      int row = ibase + m;
      int col = h * D_ + c * 16 + l16;
      float gv = acc[c][v] * sg[row * HID_ + col];
      gated[row * HID_ + col] = (__bf16)gv;
    }
  }
}

// ---------------------------------------------------------------------------
// Output GEMM: out = gated[4096,512] @ W_out + b_out  (f32 out)
// grid: (ROWS/16, 4), block 128 (4 waves x 4 N-tiles = 256 N per WG)
// ---------------------------------------------------------------------------
__global__ void k_out(const __bf16* __restrict__ gated,
                      const __bf16* __restrict__ Wout_t,
                      const float*  __restrict__ bout,
                      float* __restrict__ out) {
  const int lane = threadIdx.x & 31;
  const int wave = threadIdx.x >> 5;
  const int half = lane >> 4;
  const int l16  = lane & 15;
  const int arow = blockIdx.x * 16 + l16;
  const int nbase = blockIdx.y * 256 + wave * 64;

  v8f acc[4];
#pragma unroll
  for (int t = 0; t < 4; ++t) acc[t] = v8f_zero();

  for (int kc = 0; kc < HID_; kc += 32) {
    AFrag a;
    a.f4[0] = *(const float4*)&gated[arow * HID_ + kc + half * 16];
    a.f4[1] = *(const float4*)&gated[arow * HID_ + kc + half * 16 + 8];
#pragma unroll
    for (int t = 0; t < 4; ++t) {
      int n = nbase + t * 16 + l16;
      v16bf bm = *(const v16bf*)&Wout_t[n * HID_ + kc + half * 16];
      acc[t] = wmma_bf16(a.v, bm, acc[t]);
    }
  }

#pragma unroll
  for (int t = 0; t < 4; ++t) {
    int n = nbase + t * 16 + l16;
    float bias = bout[n];
#pragma unroll
    for (int v = 0; v < 8; ++v) {
      int m = v + 8 * half;
      out[(blockIdx.x * 16 + m) * E_ + n] = acc[t][v] + bias;
    }
  }
}

// ---------------------------------------------------------------------------
// Launch. Workspace (37 MiB):
//   Xb 8M | Wqk_t 2M | Wvg_t 2M | Wout_t 1M | q 4M | k 4M | tv_t 4M | sg 8M | gated 4M
// ---------------------------------------------------------------------------
extern "C" void kernel_launch(void* const* d_in, const int* in_sizes, int n_in,
                              void* d_out, int out_size, void* d_ws, size_t ws_size,
                              hipStream_t stream) {
  (void)in_sizes; (void)n_in; (void)out_size; (void)ws_size;

  const float* query = (const float*)d_in[0];
  const float* am    = (const float*)d_in[1];
  const unsigned char* kpm = (const unsigned char*)d_in[2];  // jax bool = 1 byte
  const float* Wqk = (const float*)d_in[3];
  const float* bqk = (const float*)d_in[4];
  const float* Wvg = (const float*)d_in[5];
  const float* bvg = (const float*)d_in[6];
  const float* Wout = (const float*)d_in[7];
  const float* bout = (const float*)d_in[8];
  float* out = (float*)d_out;

  char* ws = (char*)d_ws;
  __bf16* Xb     = (__bf16*)ws; ws += (size_t)ROWS_ * E_ * 2;
  __bf16* Wqk_t  = (__bf16*)ws; ws += (size_t)E_ * 1024 * 2;
  __bf16* Wvg_t  = (__bf16*)ws; ws += (size_t)E_ * 1024 * 2;
  __bf16* Wout_t = (__bf16*)ws; ws += (size_t)E_ * HID_ * 2;
  __bf16* q_bf   = (__bf16*)ws; ws += (size_t)ROWS_ * HID_ * 2;
  __bf16* k_bf   = (__bf16*)ws; ws += (size_t)ROWS_ * HID_ * 2;
  __bf16* tv_t   = (__bf16*)ws; ws += (size_t)ROWS_ * HID_ * 2;
  float*  sg     = (float*)ws;  ws += (size_t)ROWS_ * HID_ * 4;
  __bf16* gated  = (__bf16*)ws; ws += (size_t)ROWS_ * HID_ * 2;

  k_cvt_bf16<<<(ROWS_ * E_) / 256, 256, 0, stream>>>(query, Xb, ROWS_ * E_);
  k_transpose_bf16<<<dim3(1024 / 16, E_ / 16), dim3(16, 16), 0, stream>>>(Wqk, Wqk_t, E_, 1024);
  k_transpose_bf16<<<dim3(1024 / 16, E_ / 16), dim3(16, 16), 0, stream>>>(Wvg, Wvg_t, E_, 1024);
  k_transpose_bf16<<<dim3(E_ / 16, HID_ / 16), dim3(16, 16), 0, stream>>>(Wout, Wout_t, HID_, E_);

  k_qkvg<<<dim3(ROWS_ / 16, 8), 128, 0, stream>>>(Xb, Wqk_t, Wvg_t, bqk, bvg,
                                                  q_bf, k_bf, tv_t, sg);
  k_attn<<<dim3(ROWS_ / 16), 256, 0, stream>>>(q_bf, k_bf, tv_t, sg, am, kpm, gated);
  k_out<<<dim3(ROWS_ / 16, E_ / 256), 128, 0, stream>>>(gated, Wout_t, bout, out);
}
